// MambaCPAModelWoPretrained_73340861546558
// MI455X (gfx1250) — matrix operand order repaired and checked by reading, not compiled, not run on hardware
//
#include <hip/hip_runtime.h>
#include <math.h>

// ---------------- model dims ----------------
#define DM      256
#define DIN     512
#define NSTATE  16
#define DTRANK  16
#define NBATCH  8
#define LPROT   1024
#define LCOMP   128
#define NPGN    64
#define NNODES  512
#define NEDGES  4096
#define NHEADS  8
#define HDIM    32
#define EPSF    1e-5f

typedef __attribute__((ext_vector_type(8)))  __bf16 v8bf;
typedef __attribute__((ext_vector_type(16))) __bf16 v16bf;
typedef __attribute__((ext_vector_type(8)))  float  v8f;

// ---------------- device helpers ----------------
__device__ __forceinline__ unsigned short f2bf(float f) {
  unsigned int u = __float_as_uint(f);
  unsigned int r = (u + 0x7fffu + ((u >> 16) & 1u)) >> 16;   // RNE
  return (unsigned short)r;
}

__device__ __forceinline__ float actf(float v, int act) {
  switch (act) {
    case 1: return v / (1.f + expf(-v));                          // silu
    case 2: return 0.5f * v * (1.f + erff(v * 0.70710678118f));   // gelu (exact)
    case 3: return 1.f / (1.f + expf(-v));                        // sigmoid
    case 4: return (v > 20.f) ? v : log1pf(expf(v));              // softplus
    default: return v;
  }
}

__device__ __forceinline__ float wredsum(float v) {
  for (int o = 16; o > 0; o >>= 1) v += __shfl_xor(v, o, 32);
  return v;
}
__device__ __forceinline__ float wredmax(float v) {
  for (int o = 16; o > 0; o >>= 1) v = fmaxf(v, __shfl_xor(v, o, 32));
  return v;
}

// ---------------- kernels ----------------

// fp32 -> zero-padded bf16, optional transpose, batched via blockIdx.z
__global__ void k_convert(const float* __restrict__ src, unsigned short* __restrict__ dst,
                          int R, int C, int Rp, int Cp, int ldSrc,
                          long long sB, long long dB, int transp) {
  long long tot = (long long)Rp * Cp;
  long long idx = (long long)blockIdx.x * blockDim.x + threadIdx.x;
  if (idx >= tot) return;
  int r = (int)(idx / Cp), cc = (int)(idx % Cp);
  const float* s = src + (long long)blockIdx.z * sB;
  float v = 0.f;
  if (r < R && cc < C)
    v = transp ? s[(long long)cc * ldSrc + r] : s[(long long)r * ldSrc + cc];
  dst[(long long)blockIdx.z * dB + idx] = f2bf(v);
}

// Y[m,n] = act( sum_k A[m,k]*B[n,k] + bias[n] ).  bf16 inputs, fp32 accum via
// v_wmma_f32_16x16x32_bf16.  One wave per 16x16 output tile; 4 waves/block.
__global__ void k_gemm(const unsigned short* __restrict__ Au, const unsigned short* __restrict__ Bu,
                       const float* __restrict__ bias, float* __restrict__ Y,
                       int M, int N, int Kp, int ldY, int act,
                       long long sA, long long sB, long long sY) {
  int lane = threadIdx.x & 31;
  int wv   = threadIdx.x >> 5;
  int tilesN = (N + 15) >> 4;
  long long tile = (long long)blockIdx.x * 4 + wv;
  long long nt = (long long)((M + 15) >> 4) * tilesN;
  if (tile >= nt) return;
  int tm = (int)(tile / tilesN), tn = (int)(tile % tilesN);
  const __bf16* A = reinterpret_cast<const __bf16*>(Au) + (long long)blockIdx.z * sA;
  const __bf16* B = reinterpret_cast<const __bf16*>(Bu) + (long long)blockIdx.z * sB;
  float* Yb = Y + (long long)blockIdx.z * sY;
  int half = lane >> 4, l4 = lane & 15;
  // A layout: lane row = l4; elems 0..7 -> K=kb..kb+7, 8..15 -> K=kb+16..kb+23 (kb=half*8)
  const __bf16* ap = A + (long long)(tm * 16 + l4) * Kp + half * 8;
  // B layout: lane col = l4; elems 0..7 -> K=kb..kb+7, 8..15 -> K=kb+16..kb+23 (kb=half*8)
  const __bf16* bp = B + (long long)(tn * 16 + l4) * Kp + half * 8;
  v8f acc;
  #pragma unroll
  for (int i = 0; i < 8; i++) acc[i] = 0.f;
  for (int k = 0; k < Kp; k += 32) {
    v8bf a0 = *(const v8bf*)(ap + k);
    v8bf a1 = *(const v8bf*)(ap + k + 16);
    v8bf b0 = *(const v8bf*)(bp + k);
    v8bf b1 = *(const v8bf*)(bp + k + 16);
    v16bf av = __builtin_shufflevector(a0, a1, 0,1,2,3,4,5,6,7,8,9,10,11,12,13,14,15);
    v16bf bv = __builtin_shufflevector(b0, b1, 0,1,2,3,4,5,6,7,8,9,10,11,12,13,14,15);
    acc = __builtin_amdgcn_wmma_f32_16x16x32_bf16(false, av, false, bv, (short)0, acc, false, false);
  }
  int col = tn * 16 + l4;
  if (col < N) {
    float bb = bias ? bias[col] : 0.f;
    int mb = tm * 16 + half * 8;
    #pragma unroll
    for (int i = 0; i < 8; i++) {
      int row = mb + i;
      if (row < M) Yb[(long long)row * ldY + col] = actf(acc[i] + bb, act);
    }
  }
}

__global__ void k_embed(const int* __restrict__ ids, const float* __restrict__ emb,
                        float* __restrict__ out, long long total) {
  long long idx = (long long)blockIdx.x * blockDim.x + threadIdx.x;
  if (idx >= total) return;
  long long t = idx >> 8; int cidx = (int)(idx & 255);
  out[idx] = emb[(long long)ids[t] * DM + cidx];
}

__global__ void k_negexp(const float* __restrict__ a, float* __restrict__ o, int n) {
  int i = blockIdx.x * blockDim.x + threadIdx.x;
  if (i < n) o[i] = -expf(a[i]);
}

// depthwise causal conv (width 4) + silu.  xz is [M, 2*DIN], conv over cols 0..DIN-1.
__global__ void k_dwconv(const float* __restrict__ xz, const float* __restrict__ w,
                         const float* __restrict__ bias, float* __restrict__ out,
                         int L, long long total) {
  long long idx = (long long)blockIdx.x * blockDim.x + threadIdx.x;
  if (idx >= total) return;
  int d = (int)(idx & (DIN - 1));
  long long t = idx >> 9;
  int l = (int)(t % L);
  float acc = bias[d];
  #pragma unroll
  for (int j = 0; j < 4; j++) {
    int ls = l - 3 + j;
    if (ls >= 0) acc += w[d * 4 + j] * xz[(t - l + ls) * (2 * DIN) + d];
  }
  out[idx] = acc / (1.f + expf(-acc));
}

// selective-scan: one thread per (batch, channel); h[16] in registers.
__global__ void k_scan(const float* __restrict__ delta, const float* __restrict__ dbc,
                       const float* __restrict__ xc, const float* __restrict__ Aexp,
                       const float* __restrict__ Dv, float* __restrict__ y,
                       int L, int total) {
  int gid = blockIdx.x * blockDim.x + threadIdx.x;
  if (gid >= total) return;
  int b = gid / DIN, d = gid % DIN;
  float Ar[NSTATE], h[NSTATE];
  #pragma unroll
  for (int s = 0; s < NSTATE; s++) { Ar[s] = Aexp[d * NSTATE + s]; h[s] = 0.f; }
  float Dd = Dv[d];
  long long base = (long long)b * L;
  for (int l = 0; l < L; l++) {
    long long row = base + l;
    float dt = delta[row * DIN + d];
    float xv = xc[row * DIN + d];
    const float* bc = dbc + row * 48;   // [dt(16) | B(16) | C(16)]
    float dx = dt * xv, acc = 0.f;
    #pragma unroll
    for (int s = 0; s < NSTATE; s++) {
      float hn = expf(dt * Ar[s]) * h[s] + dx * bc[16 + s];
      h[s] = hn;
      acc += hn * bc[32 + s];
    }
    y[row * DIN + d] = acc + Dd * xv;
  }
}

__global__ void k_gate(float* __restrict__ y, const float* __restrict__ xz, long long n) {
  long long idx = (long long)blockIdx.x * blockDim.x + threadIdx.x;
  if (idx >= n) return;
  long long t = idx >> 9; int d = (int)(idx & (DIN - 1));
  float z = xz[t * (2 * DIN) + DIN + d];
  y[idx] *= z / (1.f + expf(-z));
}

__global__ void k_add2(float* __restrict__ y, const float* __restrict__ a,
                       const float* __restrict__ b, long long n) {
  long long i = (long long)blockIdx.x * blockDim.x + threadIdx.x;
  if (i < n) y[i] = a[i] + b[i];
}
__global__ void k_add3(float* __restrict__ y, const float* __restrict__ a,
                       const float* __restrict__ b, const float* __restrict__ c2, long long n) {
  long long i = (long long)blockIdx.x * blockDim.x + threadIdx.x;
  if (i < n) y[i] = a[i] + b[i] + c2[i];
}
__global__ void k_mul(float* __restrict__ y, const float* __restrict__ b, long long n) {
  long long i = (long long)blockIdx.x * blockDim.x + threadIdx.x;
  if (i < n) y[i] *= b[i];
}

__global__ void k_flip(const float* __restrict__ x, float* __restrict__ y, int L, long long total) {
  long long idx = (long long)blockIdx.x * blockDim.x + threadIdx.x;
  if (idx >= total) return;
  int cidx = (int)(idx & 255);
  long long t = idx >> 8;
  int l = (int)(t % L);
  y[idx] = x[((t - l) + (L - 1 - l)) * DM + cidx];
}

// wave-per-row layernorm, C = 256
__global__ void k_layernorm(const float* __restrict__ x, float* __restrict__ y,
                            const float* __restrict__ g, const float* __restrict__ b,
                            long long rows) {
  int lane = threadIdx.x & 31;
  long long row = (long long)blockIdx.x * 8 + (threadIdx.x >> 5);
  if (row >= rows) return;
  const float* xr = x + row * DM;
  float v[8];
  #pragma unroll
  for (int i = 0; i < 8; i++) v[i] = xr[lane * 8 + i];
  float s = 0.f;
  #pragma unroll
  for (int i = 0; i < 8; i++) s += v[i];
  float mean = wredsum(s) * (1.f / DM);
  float vs = 0.f;
  #pragma unroll
  for (int i = 0; i < 8; i++) { float d = v[i] - mean; vs += d * d; }
  float var = wredsum(vs) * (1.f / DM);
  float rs = rsqrtf(var + EPSF);
  #pragma unroll
  for (int i = 0; i < 8; i++) {
    int cidx = lane * 8 + i;
    y[row * DM + cidx] = g[cidx] * (v[i] - mean) * rs + b[cidx];
  }
}

// wave-per-row softmax over `len` with logit scale, in place
__global__ void k_softmax(float* __restrict__ S, long long rows, int len, float scale) {
  int lane = threadIdx.x & 31;
  long long row = (long long)blockIdx.x * 8 + (threadIdx.x >> 5);
  if (row >= rows) return;
  float* r = S + row * (long long)len;
  float m = -3.4e38f;
  for (int i = lane; i < len; i += 32) m = fmaxf(m, r[i]);
  m = wredmax(m);
  float s = 0.f;
  for (int i = lane; i < len; i += 32) s += expf(scale * (r[i] - m));
  s = wredsum(s);
  float inv = 1.f / s;
  for (int i = lane; i < len; i += 32) r[i] = expf(scale * (r[i] - m)) * inv;
}

__global__ void k_seqmax(const float* __restrict__ x, float* __restrict__ y, int L, int total) {
  int idx = blockIdx.x * blockDim.x + threadIdx.x;
  if (idx >= total) return;
  int b = idx >> 8, cidx = idx & 255;
  float m = -3.4e38f;
  for (int l = 0; l < L; l++) m = fmaxf(m, x[((long long)b * L + l) * DM + cidx]);
  y[idx] = m;
}

__global__ void k_edge_pre(const float* __restrict__ q, const float* __restrict__ k,
                           const float* __restrict__ v, const float* __restrict__ e,
                           const int* __restrict__ src, const int* __restrict__ dst,
                           float* __restrict__ ve, float* __restrict__ logit) {
  int ed = blockIdx.x * blockDim.x + threadIdx.x;
  if (ed >= NEDGES) return;
  int s = src[ed], d = dst[ed];
  const float* qr = q + (long long)d * DM;
  const float* kr = k + (long long)s * DM;
  const float* vr = v + (long long)s * DM;
  const float* er = e + (long long)ed * DM;
  float* vo = ve + (long long)ed * DM;
  float acc = 0.f;
  for (int cidx = 0; cidx < DM; cidx++) {
    float kk = kr[cidx] + er[cidx];
    vo[cidx] = vr[cidx] + er[cidx];
    acc += qr[cidx] * kk;
  }
  logit[ed] = acc * (1.f / 16.f);   // 1/sqrt(256)
}

__global__ void k_segstats(const float* __restrict__ logit, const int* __restrict__ dst,
                           float* __restrict__ smax, float* __restrict__ ssum) {
  int n = blockIdx.x * blockDim.x + threadIdx.x;
  if (n >= NNODES) return;
  float m = -3.4e38f;
  for (int ed = 0; ed < NEDGES; ed++) if (dst[ed] == n) m = fmaxf(m, logit[ed]);
  float s = 0.f;
  for (int ed = 0; ed < NEDGES; ed++) if (dst[ed] == n) s += expf(logit[ed] - m);
  smax[n] = m; ssum[n] = s;
}

__global__ void k_alpha(const float* __restrict__ logit, const int* __restrict__ dst,
                        const float* __restrict__ smax, const float* __restrict__ ssum,
                        float* __restrict__ alpha) {
  int ed = blockIdx.x * blockDim.x + threadIdx.x;
  if (ed >= NEDGES) return;
  int d = dst[ed];
  alpha[ed] = expf(logit[ed] - smax[d]) / (ssum[d] + 1e-16f);
}

__global__ void k_aggregate(const float* __restrict__ alpha, const float* __restrict__ ve,
                            const int* __restrict__ dst, float* __restrict__ out) {
  int gid = blockIdx.x * blockDim.x + threadIdx.x;
  if (gid >= NNODES * DM) return;
  int n = gid >> 8, cidx = gid & 255;
  float acc = 0.f;
  for (int ed = 0; ed < NEDGES; ed++)
    if (dst[ed] == n) acc += alpha[ed] * ve[(long long)ed * DM + cidx];
  out[gid] = acc;
}

__global__ void k_beta(const float* __restrict__ outn, const float* __restrict__ xrn,
                       const float* __restrict__ bw, float* __restrict__ beta) {
  int n = blockIdx.x * blockDim.x + threadIdx.x;
  if (n >= NNODES) return;
  float acc = 0.f;
  for (int cidx = 0; cidx < DM; cidx++) {
    float o = outn[(long long)n * DM + cidx], xr = xrn[(long long)n * DM + cidx];
    acc += o * bw[cidx] + xr * bw[DM + cidx] + (o - xr) * bw[2 * DM + cidx];
  }
  beta[n] = 1.f / (1.f + expf(-acc));
}

__global__ void k_combine(const float* __restrict__ beta, const float* __restrict__ xrn,
                          const float* __restrict__ outn, float* __restrict__ h) {
  int gid = blockIdx.x * blockDim.x + threadIdx.x;
  if (gid >= NNODES * DM) return;
  float be = beta[gid >> 8];
  h[gid] = be * xrn[gid] + (1.f - be) * outn[gid];
}

__global__ void k_graphnorm(const float* __restrict__ x, float* __restrict__ y,
                            const float* __restrict__ g, const float* __restrict__ bb,
                            const float* __restrict__ a) {
  int gid = blockIdx.x * blockDim.x + threadIdx.x;
  if (gid >= NBATCH * DM) return;
  int b = gid >> 8, cidx = gid & 255;
  const float* xb = x + (long long)b * NPGN * DM + cidx;
  float mean = 0.f;
  for (int i = 0; i < NPGN; i++) mean += xb[i * DM];
  mean *= (1.f / NPGN);
  float am = a[cidx] * mean;
  float var = 0.f;
  for (int i = 0; i < NPGN; i++) { float s = xb[i * DM] - am; var += s * s; }
  var *= (1.f / NPGN);
  float rs = rsqrtf(var + EPSF);
  for (int i = 0; i < NPGN; i++) {
    float s = xb[i * DM] - am;
    y[(long long)b * NPGN * DM + i * DM + cidx] = g[cidx] * s * rs + bb[cidx];
  }
}

__global__ void k_segmaxb(const float* __restrict__ x, const int* __restrict__ bv,
                          float* __restrict__ mol) {
  int gid = blockIdx.x * blockDim.x + threadIdx.x;
  if (gid >= NBATCH * DM) return;
  int b = gid >> 8, cidx = gid & 255;
  float m = -3.4e38f;
  for (int n = 0; n < NNODES; n++)
    if (bv[n] == b) m = fmaxf(m, x[(long long)n * DM + cidx]);
  mol[gid] = m;
}

__global__ void k_bn(const float* __restrict__ x, float* __restrict__ y,
                     const float* __restrict__ g, const float* __restrict__ b, int M, int C) {
  int cidx = blockIdx.x * blockDim.x + threadIdx.x;
  if (cidx >= C) return;
  float mean = 0.f;
  for (int m = 0; m < M; m++) mean += x[(long long)m * C + cidx];
  mean /= M;
  float var = 0.f;
  for (int m = 0; m < M; m++) { float d = x[(long long)m * C + cidx] - mean; var += d * d; }
  var /= M;
  float rs = rsqrtf(var + EPSF);
  for (int m = 0; m < M; m++)
    y[(long long)m * C + cidx] = g[cidx] * (x[(long long)m * C + cidx] - mean) * rs + b[cidx];
}

// dst[m*ld + c] = a*w (or a), dst[m*ld + C + c] = b*(1-w) (or b)
__global__ void k_pack(const float* __restrict__ a, const float* __restrict__ b,
                       const float* __restrict__ w, float* __restrict__ dst,
                       int ld, int M, int C) {
  int gid = blockIdx.x * blockDim.x + threadIdx.x;
  if (gid >= M * C) return;
  int m = gid / C, cidx = gid % C;
  float av = a[(long long)m * C + cidx], bv2 = b[(long long)m * C + cidx];
  if (w) {
    float wv = w[(long long)m * C + cidx];
    dst[(long long)m * ld + cidx] = av * wv;
    dst[(long long)m * ld + C + cidx] = bv2 * (1.f - wv);
  } else {
    dst[(long long)m * ld + cidx] = av;
    dst[(long long)m * ld + C + cidx] = bv2;
  }
}

// ---------------- host side ----------------
namespace {

struct PC { void* const* din; int i;
  const float* f() { return (const float*)din[i++]; } };

struct MambaCore { const float *A_log,*D,*conv_b,*conv_w,*dt_b,*dt_w,*xproj_w; };
struct MambaIO   { const float *in_b,*in_w,*out_b,*out_w; };
struct Bidir     { MambaCore bwd, fwd; const float *ln_b,*ln_g; MambaIO sh; };
struct Swi       { const float *b1,*b2,*b3,*w1,*w2,*w3; };
struct AmL       { const float *in_b,*in_w; Bidir mm; const float *n1_b,*n1_g,*out_b,*out_w; };
struct FF        { const float *l1_b,*l1_w,*l2_b,*l2_w; };
struct GpsP      { Bidir attn; const float *beta_w,*bn_a,*bn_b,*bn_g,*e_w,*k_b,*k_w,*lin_b,*lin_w;
                   Swi mlp; const float *q_b,*q_w,*skip_b,*skip_w,*v_b,*v_w; };
struct MambaFull { MambaCore co; MambaIO io; };
struct MmgP      { FF ff1, ff2; Swi gg, gp, gs; };
struct PredP     { const float *bn1_b,*bn1_g,*bn2_b,*bn2_g,*bn3_b,*bn3_g,
                   *l1_b,*l1_w,*l2_b,*l2_w,*l3_b,*l3_w,*l4_b,*l4_w; };

// all loaders consume leaves in JAX pytree (sorted-dict-key) order
static MambaCore ldCore(PC& c){ MambaCore m; m.A_log=c.f(); m.D=c.f(); m.conv_b=c.f();
  m.conv_w=c.f(); m.dt_b=c.f(); m.dt_w=c.f(); m.xproj_w=c.f(); return m; }
static MambaIO ldIO(PC& c){ MambaIO m; m.in_b=c.f(); m.in_w=c.f(); m.out_b=c.f(); m.out_w=c.f(); return m; }
static Bidir ldBidir(PC& c){ Bidir b; b.bwd=ldCore(c); b.fwd=ldCore(c); b.ln_b=c.f(); b.ln_g=c.f();
  b.sh=ldIO(c); return b; }
static Swi ldSwi(PC& c){ Swi s; s.b1=c.f(); s.b2=c.f(); s.b3=c.f(); s.w1=c.f(); s.w2=c.f(); s.w3=c.f(); return s; }
static AmL ldAm(PC& c){ AmL a; a.in_b=c.f(); a.in_w=c.f(); a.mm=ldBidir(c); a.n1_b=c.f(); a.n1_g=c.f();
  a.out_b=c.f(); a.out_w=c.f(); return a; }
static FF ldFF(PC& c){ FF f; f.l1_b=c.f(); f.l1_w=c.f(); f.l2_b=c.f(); f.l2_w=c.f(); return f; }
static GpsP ldGps(PC& c){ GpsP g; g.attn=ldBidir(c); g.beta_w=c.f(); g.bn_a=c.f(); g.bn_b=c.f();
  g.bn_g=c.f(); g.e_w=c.f(); g.k_b=c.f(); g.k_w=c.f(); g.lin_b=c.f(); g.lin_w=c.f();
  g.mlp=ldSwi(c); g.q_b=c.f(); g.q_w=c.f(); g.skip_b=c.f(); g.skip_w=c.f(); g.v_b=c.f(); g.v_w=c.f(); return g; }
static MambaFull ldMF(PC& c){ MambaFull m;
  m.co.A_log=c.f(); m.co.D=c.f(); m.co.conv_b=c.f(); m.co.conv_w=c.f();
  m.co.dt_b=c.f(); m.co.dt_w=c.f();
  m.io.in_b=c.f(); m.io.in_w=c.f(); m.io.out_b=c.f(); m.io.out_w=c.f();
  m.co.xproj_w=c.f(); return m; }
static PredP ldPred(PC& c){ PredP p; p.bn1_b=c.f(); p.bn1_g=c.f(); p.bn2_b=c.f(); p.bn2_g=c.f();
  p.bn3_b=c.f(); p.bn3_g=c.f(); p.l1_b=c.f(); p.l1_w=c.f(); p.l2_b=c.f(); p.l2_w=c.f();
  p.l3_b=c.f(); p.l3_w=c.f(); p.l4_b=c.f(); p.l4_w=c.f(); return p; }

struct Ctx {
  hipStream_t st;
  unsigned short *GA, *GB;
  float *T0,*T1,*T2,*T3,*T4,*T5,*T6,*T7,*S,*AEXP;
};

static inline long long cdiv(long long a, long long b){ return (a + b - 1) / b; }
static inline int rup(int x, int m){ return ((x + m - 1) / m) * m; }

// Y[M,N](ldY) = act(X[M,K](ldX) @ W[N,K](ldW)^T + bias)
static void gemm(const Ctx& c, const float* X, int ldX, const float* W, int ldW,
                 const float* bias, float* Y, int ldY, int M, int N, int K, int act) {
  int Kp = rup(K, 32), Mp = rup(M, 16), Np = rup(N, 16);
  long long ta = (long long)Mp * Kp;
  k_convert<<<dim3((unsigned)cdiv(ta,256),1,1),256,0,c.st>>>(X, c.GA, M, K, Mp, Kp, ldX, 0LL, 0LL, 0);
  long long tb = (long long)Np * Kp;
  k_convert<<<dim3((unsigned)cdiv(tb,256),1,1),256,0,c.st>>>(W, c.GB, N, K, Np, Kp, ldW, 0LL, 0LL, 0);
  long long tiles = (long long)(Mp / 16) * (Np / 16);
  k_gemm<<<dim3((unsigned)cdiv(tiles,4),1,1),128,0,c.st>>>(c.GA, c.GB, bias, Y, M, N, Kp, ldY, act, 0LL, 0LL, 0LL);
}

static void run_swiglu(const Ctx& c, const Swi& s, const float* x, float* y, int M) {
  gemm(c, x, DM, s.w1, DM, s.b1, c.T0, 4*DM, M, 4*DM, DM, 1);
  gemm(c, x, DM, s.w2, DM, s.b2, c.T1, 4*DM, M, 4*DM, DM, 0);
  long long n = (long long)M * 4 * DM;
  k_mul<<<dim3((unsigned)cdiv(n,256)),256,0,c.st>>>(c.T0, c.T1, n);
  gemm(c, c.T0, 4*DM, s.w3, 4*DM, s.b3, y, DM, M, DM, 4*DM, 0);
}

static void run_mamba(const Ctx& c, const MambaIO& io, const MambaCore& co,
                      const float* x, float* y, int Bn, int L) {
  long long M = (long long)Bn * L;
  k_negexp<<<dim3((unsigned)cdiv(DIN*NSTATE,256)),256,0,c.st>>>(co.A_log, c.AEXP, DIN*NSTATE);
  gemm(c, x, DM, io.in_w, DM, io.in_b, c.T0, 2*DIN, (int)M, 2*DIN, DM, 0);
  long long md = M * DIN;
  k_dwconv<<<dim3((unsigned)cdiv(md,256)),256,0,c.st>>>(c.T0, co.conv_w, co.conv_b, c.T1, L, md);
  gemm(c, c.T1, DIN, co.xproj_w, DIN, nullptr, c.T4, 48, (int)M, 48, DIN, 0);
  gemm(c, c.T4, 48, co.dt_w, DTRANK, co.dt_b, c.T2, DIN, (int)M, DIN, DTRANK, 4);
  k_scan<<<dim3((unsigned)cdiv(Bn*DIN,256)),256,0,c.st>>>(c.T2, c.T4, c.T1, c.AEXP, co.D, c.T3, L, Bn*DIN);
  k_gate<<<dim3((unsigned)cdiv(md,256)),256,0,c.st>>>(c.T3, c.T0, md);
  gemm(c, c.T3, DIN, io.out_w, DIN, io.out_b, y, DM, (int)M, DM, DIN, 0);
}

static void run_bidir(const Ctx& c, const Bidir& bd, const float* x, float* y, int Bn, int L) {
  long long M = (long long)Bn * L, n = M * DM;
  run_mamba(c, bd.sh, bd.fwd, x, c.T5, Bn, L);
  k_flip<<<dim3((unsigned)cdiv(n,256)),256,0,c.st>>>(x, c.T6, L, n);
  run_mamba(c, bd.sh, bd.bwd, c.T6, c.T7, Bn, L);
  k_flip<<<dim3((unsigned)cdiv(n,256)),256,0,c.st>>>(c.T7, c.T6, L, n);
  k_add3<<<dim3((unsigned)cdiv(n,256)),256,0,c.st>>>(c.T5, x, c.T5, c.T6, n);
  k_layernorm<<<dim3((unsigned)cdiv(M,8)),256,0,c.st>>>(c.T5, y, bd.ln_g, bd.ln_b, M);
}

static void run_mha(const Ctx& c, const AmL& a, float* x, int Bn, int L) {
  long long M = (long long)Bn * L, n = M * DM;
  gemm(c, x, DM, a.in_w, DM, a.in_b, c.T0, 3*DM, (int)M, 3*DM, DM, 0);
  const float scale = 0.1767766953f;   // 1/sqrt(32)
  for (int b = 0; b < Bn; b++) {
    const float* qkvb = c.T0 + (long long)b * L * 3 * DM;
    long long qh = (long long)L * HDIM;
    k_convert<<<dim3((unsigned)cdiv(qh,256),1,NHEADS),256,0,c.st>>>(qkvb, c.GA, L, HDIM, L, HDIM, 3*DM, 32LL, qh, 0);
    k_convert<<<dim3((unsigned)cdiv(qh,256),1,NHEADS),256,0,c.st>>>(qkvb + DM, c.GB, L, HDIM, L, HDIM, 3*DM, 32LL, qh, 0);
    long long tiles = (long long)(L/16) * (L/16);
    k_gemm<<<dim3((unsigned)cdiv(tiles,4),1,NHEADS),128,0,c.st>>>(c.GA, c.GB, nullptr, c.S,
        L, L, 32, L, 0, qh, qh, (long long)L * L);
    long long rows = (long long)NHEADS * L;
    k_softmax<<<dim3((unsigned)cdiv(rows,8)),256,0,c.st>>>(c.S, rows, L, scale);
    long long pt = (long long)L * L;
    k_convert<<<dim3((unsigned)cdiv(pt,256),1,NHEADS),256,0,c.st>>>(c.S, c.GA, L, L, L, L, L, pt, pt, 0);
    long long vt = (long long)HDIM * L;
    k_convert<<<dim3((unsigned)cdiv(vt,256),1,NHEADS),256,0,c.st>>>(qkvb + 2*DM, c.GB, HDIM, L, HDIM, L, 3*DM, 32LL, vt, 1);
    long long tiles2 = (long long)(L/16) * (HDIM/16);
    k_gemm<<<dim3((unsigned)cdiv(tiles2,4),1,NHEADS),128,0,c.st>>>(c.GA, c.GB, nullptr,
        c.T7 + (long long)b * L * DM, L, HDIM, L, DM, 0, pt, vt, (long long)HDIM);
  }
  gemm(c, c.T7, DM, a.out_w, DM, a.out_b, c.T6, DM, (int)M, DM, DM, 0);
  k_add2<<<dim3((unsigned)cdiv(n,256)),256,0,c.st>>>(c.T6, c.T6, x, n);
  k_layernorm<<<dim3((unsigned)cdiv(M,8)),256,0,c.st>>>(c.T6, x, a.n1_g, a.n1_b, M);
}

static void ff_fusion(const Ctx& c, const FF& f, const float* fd, const float* fp,
                      float* dst, int dstLd, float* CAT, float* W1, float* W2) {
  unsigned g = (unsigned)cdiv(NBATCH * DM, 256);
  k_pack<<<dim3(g),256,0,c.st>>>(fd, fp, (const float*)nullptr, CAT, 2*DM, NBATCH, DM);
  gemm(c, CAT, 2*DM, f.l1_w, 2*DM, f.l1_b, c.T0, 4*DM, NBATCH, 4*DM, 2*DM, 2);
  gemm(c, c.T0, 4*DM, f.l2_w, 4*DM, f.l2_b, W1, DM, NBATCH, DM, 4*DM, 3);
  k_pack<<<dim3(g),256,0,c.st>>>(fd, fp, W1, CAT, 2*DM, NBATCH, DM);
  gemm(c, CAT, 2*DM, f.l1_w, 2*DM, f.l1_b, c.T0, 4*DM, NBATCH, 4*DM, 2*DM, 2);
  gemm(c, c.T0, 4*DM, f.l2_w, 4*DM, f.l2_b, W2, DM, NBATCH, DM, 4*DM, 3);
  k_pack<<<dim3(g),256,0,c.st>>>(fd, fp, W2, dst, dstLd, NBATCH, DM);
}

} // namespace

extern "C" void kernel_launch(void* const* d_in, const int* in_sizes, int n_in,
                              void* d_out, int out_size, void* d_ws, size_t ws_size,
                              hipStream_t stream) {
  (void)in_sizes; (void)n_in; (void)out_size; (void)ws_size;
  const int*   pro_ids   = (const int*)d_in[0];
  const int*   smi_ids   = (const int*)d_in[1];
  const float* node_x    = (const float*)d_in[2];
  const int*   edge_index= (const int*)d_in[3];
  const float* edge_attr = (const float*)d_in[4];
  const int*   batch_vec = (const int*)d_in[5];

  PC pc{d_in, 6};
  AmL comp_am[2] = { ldAm(pc), ldAm(pc) };
  const float* comp_emb = pc.f();
  GpsP gps = ldGps(pc);
  MmgP mg; mg.ff1 = ldFF(pc); mg.ff2 = ldFF(pc); mg.gg = ldSwi(pc); mg.gp = ldSwi(pc); mg.gs = ldSwi(pc);
  MambaFull pre_comp = ldMF(pc);
  MambaFull pre_prot = ldMF(pc);
  PredP pr = ldPred(pc);
  AmL prot_am[2] = { ldAm(pc), ldAm(pc) };
  const float* prot_emb = pc.f();
  const float* x2h_b = pc.f();
  const float* x2h_w = pc.f();

  char* base = (char*)d_ws; size_t off = 0;
  auto alloc = [&](size_t bytes) -> void* {
    void* p = base + off; off += (bytes + 255) & ~(size_t)255; return p;
  };

  Ctx c; c.st = stream;
  c.T0 = (float*)alloc(sizeof(float) * (size_t)8192 * 1024);
  c.T1 = (float*)alloc(sizeof(float) * (size_t)8192 * 512);
  c.T2 = (float*)alloc(sizeof(float) * (size_t)8192 * 512);
  c.T3 = (float*)alloc(sizeof(float) * (size_t)8192 * 512);
  c.T4 = (float*)alloc(sizeof(float) * (size_t)8192 * 48);
  c.T5 = (float*)alloc(sizeof(float) * (size_t)8192 * 256);
  c.T6 = (float*)alloc(sizeof(float) * (size_t)8192 * 256);
  c.T7 = (float*)alloc(sizeof(float) * (size_t)8192 * 256);
  c.S  = (float*)alloc(sizeof(float) * (size_t)NHEADS * 1024 * 1024);
  c.GA = (unsigned short*)alloc(sizeof(unsigned short) * (size_t)10 * 1024 * 1024);
  c.GB = (unsigned short*)alloc(sizeof(unsigned short) * (size_t)2 * 1024 * 1024);
  c.AEXP = (float*)alloc(sizeof(float) * DIN * NSTATE);

  float* XN   = (float*)alloc(sizeof(float) * NNODES * DM);
  float* QN   = (float*)alloc(sizeof(float) * NNODES * DM);
  float* KN   = (float*)alloc(sizeof(float) * NNODES * DM);
  float* VN   = (float*)alloc(sizeof(float) * NNODES * DM);
  float* OUTN = (float*)alloc(sizeof(float) * NNODES * DM);
  float* XRN  = (float*)alloc(sizeof(float) * NNODES * DM);
  float* HN   = (float*)alloc(sizeof(float) * NNODES * DM);
  float* H2N  = (float*)alloc(sizeof(float) * NNODES * DM);
  float* HDN  = (float*)alloc(sizeof(float) * NNODES * DM);
  float* EN   = (float*)alloc(sizeof(float) * NEDGES * DM);
  float* VE   = (float*)alloc(sizeof(float) * NEDGES * DM);
  float* LOGIT= (float*)alloc(sizeof(float) * NEDGES);
  float* ALPHA= (float*)alloc(sizeof(float) * NEDGES);
  float* SMAX = (float*)alloc(sizeof(float) * NNODES);
  float* SSUM = (float*)alloc(sizeof(float) * NNODES);
  float* BETA = (float*)alloc(sizeof(float) * NNODES);
  float* PROT = (float*)alloc(sizeof(float) * (size_t)NBATCH * LPROT * DM);
  float* COMP = (float*)alloc(sizeof(float) * (size_t)NBATCH * LCOMP * DM);
  float* MOL  = (float*)alloc(sizeof(float) * NBATCH * DM);
  float* MPb  = (float*)alloc(sizeof(float) * NBATCH * DM);
  float* MSb  = (float*)alloc(sizeof(float) * NBATCH * DM);
  float* MG2  = (float*)alloc(sizeof(float) * NBATCH * DM);
  float* MS2  = (float*)alloc(sizeof(float) * NBATCH * DM);
  float* MP2  = (float*)alloc(sizeof(float) * NBATCH * DM);
  float* W1b  = (float*)alloc(sizeof(float) * NBATCH * DM);
  float* W2b  = (float*)alloc(sizeof(float) * NBATCH * DM);
  float* CAT  = (float*)alloc(sizeof(float) * NBATCH * 2 * DM);
  float* CP   = (float*)alloc(sizeof(float) * NBATCH * 4 * DM);
  float* PB0  = (float*)alloc(sizeof(float) * NBATCH * 1024);
  float* PB1  = (float*)alloc(sizeof(float) * NBATCH * 1024);

  const int* esrc = edge_index;
  const int* edst = edge_index + NEDGES;
  long long nNode = (long long)NNODES * DM;
  unsigned gN = (unsigned)cdiv(nNode, 256);

  // ---- GPS conv ----
  gemm(c, node_x, 26, x2h_w, 26, x2h_b, XN, DM, NNODES, DM, 26, 0);
  gemm(c, edge_attr, 14, gps.e_w, 14, nullptr, EN, DM, NEDGES, DM, 14, 0);
  gemm(c, XN, DM, gps.q_w, DM, gps.q_b, QN, DM, NNODES, DM, DM, 0);
  gemm(c, XN, DM, gps.k_w, DM, gps.k_b, KN, DM, NNODES, DM, DM, 0);
  gemm(c, XN, DM, gps.v_w, DM, gps.v_b, VN, DM, NNODES, DM, DM, 0);
  k_edge_pre<<<dim3((unsigned)cdiv(NEDGES,64)),64,0,stream>>>(QN, KN, VN, EN, esrc, edst, VE, LOGIT);
  k_segstats<<<dim3((unsigned)cdiv(NNODES,256)),256,0,stream>>>(LOGIT, edst, SMAX, SSUM);
  k_alpha<<<dim3((unsigned)cdiv(NEDGES,256)),256,0,stream>>>(LOGIT, edst, SMAX, SSUM, ALPHA);
  k_aggregate<<<dim3(gN),256,0,stream>>>(ALPHA, VE, edst, OUTN);
  gemm(c, XN, DM, gps.skip_w, DM, gps.skip_b, XRN, DM, NNODES, DM, DM, 0);
  k_beta<<<dim3((unsigned)cdiv(NNODES,64)),64,0,stream>>>(OUTN, XRN, gps.beta_w, BETA);
  k_combine<<<dim3(gN),256,0,stream>>>(BETA, XRN, OUTN, HN);
  gemm(c, HN, DM, gps.lin_w, DM, gps.lin_b, H2N, DM, NNODES, DM, DM, 0);
  k_add2<<<dim3(gN),256,0,stream>>>(H2N, H2N, XN, nNode);
  k_graphnorm<<<dim3((unsigned)cdiv(NBATCH*DM,256)),256,0,stream>>>(H2N, HN, gps.bn_g, gps.bn_b, gps.bn_a);
  run_bidir(c, gps.attn, XN, HDN, NBATCH, NPGN);
  k_add2<<<dim3(gN),256,0,stream>>>(HDN, HDN, XN, nNode);
  k_graphnorm<<<dim3((unsigned)cdiv(NBATCH*DM,256)),256,0,stream>>>(HDN, HDN, gps.bn_g, gps.bn_b, gps.bn_a);
  k_add2<<<dim3(gN),256,0,stream>>>(OUTN, HN, HDN, nNode);
  run_swiglu(c, gps.mlp, OUTN, H2N, NNODES);
  k_add2<<<dim3(gN),256,0,stream>>>(OUTN, OUTN, H2N, nNode);
  k_graphnorm<<<dim3((unsigned)cdiv(NBATCH*DM,256)),256,0,stream>>>(OUTN, XN, gps.bn_g, gps.bn_b, gps.bn_a);
  k_segmaxb<<<dim3((unsigned)cdiv(NBATCH*DM,256)),256,0,stream>>>(XN, batch_vec, MOL);

  // ---- sequence branches ----
  long long nprot = (long long)NBATCH * LPROT * DM;
  long long ncomp = (long long)NBATCH * LCOMP * DM;
  k_embed<<<dim3((unsigned)cdiv(nprot,256)),256,0,stream>>>(pro_ids, prot_emb, PROT, nprot);
  k_embed<<<dim3((unsigned)cdiv(ncomp,256)),256,0,stream>>>(smi_ids, comp_emb, COMP, ncomp);
  run_mamba(c, pre_prot.io, pre_prot.co, PROT, PROT, NBATCH, LPROT);
  run_mamba(c, pre_comp.io, pre_comp.co, COMP, COMP, NBATCH, LCOMP);
  for (int i = 0; i < 2; i++) {
    run_mha(c, prot_am[i], PROT, NBATCH, LPROT);
    run_bidir(c, prot_am[i].mm, PROT, PROT, NBATCH, LPROT);
  }
  for (int i = 0; i < 2; i++) {
    run_mha(c, comp_am[i], COMP, NBATCH, LCOMP);
    run_bidir(c, comp_am[i].mm, COMP, COMP, NBATCH, LCOMP);
  }
  k_seqmax<<<dim3((unsigned)cdiv(NBATCH*DM,256)),256,0,stream>>>(PROT, MPb, LPROT, NBATCH*DM);
  k_seqmax<<<dim3((unsigned)cdiv(NBATCH*DM,256)),256,0,stream>>>(COMP, MSb, LCOMP, NBATCH*DM);

  // ---- multimodal gating + fusion ----
  run_swiglu(c, mg.gg, MOL, MG2, NBATCH);
  run_swiglu(c, mg.gs, MSb, MS2, NBATCH);
  run_swiglu(c, mg.gp, MPb, MP2, NBATCH);
  ff_fusion(c, mg.ff1, MG2, MP2, CP,          4*DM, CAT, W1b, W2b);
  ff_fusion(c, mg.ff2, MS2, MP2, CP + 2*DM,   4*DM, CAT, W1b, W2b);

  // ---- prediction head ----
  k_bn<<<dim3((unsigned)cdiv(1024,256)),256,0,stream>>>(CP, PB0, pr.bn1_g, pr.bn1_b, NBATCH, 1024);
  gemm(c, PB0, 1024, pr.l1_w, 1024, pr.l1_b, PB1, 1024, NBATCH, 1024, 1024, 2);
  k_bn<<<dim3((unsigned)cdiv(1024,256)),256,0,stream>>>(PB1, PB0, pr.bn2_g, pr.bn2_b, NBATCH, 1024);
  gemm(c, PB0, 1024, pr.l2_w, 1024, pr.l2_b, PB1, 1024, NBATCH, 1024, 1024, 2);
  k_bn<<<dim3((unsigned)cdiv(1024,256)),256,0,stream>>>(PB1, PB0, pr.bn3_g, pr.bn3_b, NBATCH, 1024);
  gemm(c, PB0, 1024, pr.l3_w, 1024, pr.l3_b, PB1, 512, NBATCH, 512, 1024, 2);
  gemm(c, PB1, 512, pr.l4_w, 512, pr.l4_b, (float*)d_out, 1, NBATCH, 1, 512, 0);
}
